// MTTTMultiHeadSelfAttention_49950469652781
// MI455X (gfx1250) — matrix operand order repaired and checked by reading, not compiled, not run on hardware
//
#include <hip/hip_runtime.h>

// Problem constants (from reference): B=8, H=16, N=2048, D=64, HID=256, NIT=4
#define BB    8
#define HH    16
#define NN    2048
#define DD    64
#define HIDN  256
#define NITER 4
#define CHUNK 512            // N / NIT
#define MT    32             // M-tile rows per pipeline pass

typedef _Float16 v16h __attribute__((ext_vector_type(16)));
typedef _Float16 v8h  __attribute__((ext_vector_type(8)));
typedef float    v8f  __attribute__((ext_vector_type(8)));

__device__ __forceinline__ v8f wmma_f16(v16h a, v16h b, v8f c) {
  // D = A(16x32 f16) * B(32x16 f16) + C(16x16 f32)
  return __builtin_amdgcn_wmma_f32_16x16x32_f16(false, a, false, b, (short)0, c,
                                                false, false);
}

// A fragment (16x32 f16): lane holds row = lane%16; lanes 0-15 take K {0..7,16..23},
// lanes 16-31 take K {8..15,24..31}. base is row-major with k contiguous (ld halfs).
__device__ __forceinline__ v16h frag_a(const _Float16* base, int ld, int row0,
                                       int k0, int lane) {
  int r  = row0 + (lane & 15);
  int kb = k0 + ((lane >> 4) << 3);
  v8h lo = *(const v8h*)(base + r * ld + kb);
  v8h hi = *(const v8h*)(base + r * ld + kb + 16);
  return __builtin_shufflevector(lo, hi, 0, 1, 2, 3, 4, 5, 6, 7, 8, 9, 10, 11,
                                 12, 13, 14, 15);
}

// B fragment (32x16 f16): lane holds col n = lane%16; lanes 0-15 K {0..7,16..23},
// lanes 16-31 K {8..15,24..31}. baseT stores element (k,n) at baseT[n*ldT + k].
__device__ __forceinline__ v16h frag_b(const _Float16* baseT, int ldT, int k0,
                                       int n0, int lane) {
  int n  = n0 + (lane & 15);
  int kb = k0 + ((lane >> 4) << 3);
  v8h lo = *(const v8h*)(baseT + n * ldT + kb);
  v8h hi = *(const v8h*)(baseT + n * ldT + kb + 16);
  return __builtin_shufflevector(lo, hi, 0, 1, 2, 3, 4, 5, 6, 7, 8, 9, 10, 11,
                                 12, 13, 14, 15);
}

// Branch-free tanh using hardware v_exp_f32 / v_rcp_f32.
// tanh(u) = 1 - 2/(exp(2u)+1); exp2 saturates (inf / 0) so extremes give +-1.
__device__ __forceinline__ float tanh_fast(float u) {
  float e = __builtin_amdgcn_exp2f(u * 2.885390081777927f);  // 2*log2(e)
  return 1.0f - 2.0f * __builtin_amdgcn_rcpf(e + 1.0f);
}

// jax.nn.gelu (approximate=True) and its derivative — straight-line VALU
__device__ __forceinline__ void gelu_fwd_bwd(float x, float& g, float& gp) {
  const float C1 = 0.7978845608028654f;  // sqrt(2/pi)
  const float C2 = 0.044715f;
  float x2 = x * x;
  float u  = C1 * x * (1.0f + C2 * x2);
  float t  = tanh_fast(u);
  g  = 0.5f * x * (1.0f + t);
  gp = 0.5f * (1.0f + t) + 0.5f * x * (1.0f - t * t) * C1 * (1.0f + 3.0f * C2 * x2);
}

__global__ __launch_bounds__(256) void ttt_mhsa_kernel(
    const float* __restrict__ gq, const float* __restrict__ gk,
    const float* __restrict__ gv, const float* __restrict__ gW1,
    const float* __restrict__ gW2, float* __restrict__ gout,
    float* __restrict__ ws) {
  extern __shared__ _Float16 smem[];
  _Float16* w1t16 = smem;                  // [HIDN][DD]  W1^T (f16 operand copy)
  _Float16* w2t16 = w1t16 + HIDN * DD;     // [DD][HIDN]  W2^T
  _Float16* w2c16 = w2t16 + DD * HIDN;     // [HIDN][DD]  W2 (math layout, for bwd)
  _Float16* sX    = w2c16 + HIDN * DD;     // [MT][DD]    X tile row-major
  _Float16* sXT   = sX + MT * DD;          // [DD][MT]    X tile transposed
  _Float16* sHg   = sXT + DD * MT;         // [MT][HIDN]  gelu(h) row-major
  _Float16* sHgT  = sHg + MT * HIDN;       // [HIDN][MT]  gelu(h) transposed
  _Float16* sGp   = sHgT + HIDN * MT;      // [MT][HIDN]  gelu'(h)
  _Float16* sdO   = sGp + MT * HIDN;       // [MT][DD]    dOut row-major
  _Float16* sdOT  = sdO + MT * DD;         // [DD][MT]    dOut transposed
  _Float16* sdHT  = sdOT + DD * MT;        // [HIDN][MT]  dHpre transposed

  const int tid  = threadIdx.x;
  const int lane = tid & 31;
  const int wv   = tid >> 5;
  const int wg   = blockIdx.x;  // 0..127 = b*16 + h
  const int b    = wg >> 4;
  const int h    = wg & 15;

  // f32 master weights in workspace: per-WG 32768 floats
  float* m1 = ws + (size_t)wg * (size_t)(2 * HIDN * DD);  // [HIDN][DD] = W1^T
  float* m2 = m1 + HIDN * DD;                             // [DD][HIDN] = W2^T

  // init masters from global W1 (H,D,HID) / W2 (H,HID,D), transposed
  const float* W1h = gW1 + (size_t)h * DD * HIDN;
  const float* W2h = gW2 + (size_t)h * HIDN * DD;
  for (int i = tid; i < HIDN * DD; i += 256) {
    int e = i >> 6, d = i & (DD - 1);
    m1[i] = W1h[d * HIDN + e];
  }
  for (int i = tid; i < DD * HIDN; i += 256) {
    int o = i >> 8, e = i & (HIDN - 1);
    m2[i] = W2h[e * DD + o];
  }
  __threadfence_block();
  __syncthreads();

  const float GS = 1.0f / (float)(CHUNK * DD);  // lr * dLoss/dOut scale = 1/32768

  const float* kb_ = gk + (size_t)wg * NN * DD;
  const float* vb_ = gv + (size_t)wg * NN * DD;

  for (int it = 0; it < NITER; ++it) {
    // refresh f16 operand copies from f32 masters (step-start weights)
    for (int i = tid; i < HIDN * DD; i += 256) w1t16[i] = (_Float16)m1[i];
    for (int i = tid; i < DD * HIDN; i += 256) w2t16[i] = (_Float16)m2[i];
    for (int i = tid; i < HIDN * DD; i += 256) {
      int e = i >> 6, o = i & (DD - 1);
      w2c16[i] = (_Float16)m2[o * HIDN + e];
    }
    __syncthreads();

    const float* xg0 = kb_ + (size_t)it * CHUNK * DD;
    const float* yg0 = vb_ + (size_t)it * CHUNK * DD;

#pragma unroll 1
    for (int mt = 0; mt < CHUNK / MT; ++mt) {
      const float* xg = xg0 + mt * MT * DD;
      const float* yg = yg0 + mt * MT * DD;

      // Phase 1: stage X tile (both layouts, f16); prefetch next tile into GL2/L0
      for (int i = tid; i < MT * DD; i += 256) {
        int m = i >> 6, d = i & (DD - 1);
        _Float16 xv = (_Float16)xg[i];
        sX[m * DD + d]  = xv;
        sXT[d * MT + m] = xv;
      }
      if (mt + 1 < CHUNK / MT) {
        __builtin_prefetch(xg + MT * DD + tid * 8, 0, 0);
        __builtin_prefetch(yg + MT * DD + tid * 8, 0, 0);
      }
      __syncthreads();

      // Phase 2: Hpre = X @ W1 ; Hg = gelu(Hpre), keep gelu'
      for (int t = wv; t < 2 * (HIDN / 16); t += 8) {
        int m16 = t >> 4, et = t & 15;
        v8f c = {};
#pragma unroll
        for (int kk = 0; kk < DD; kk += 32) {
          v16h a  = frag_a(sX, DD, m16 * 16, kk, lane);
          v16h bm = frag_b(w1t16, DD, kk, et * 16, lane);
          c = wmma_f16(a, bm, c);
        }
        int col = et * 16 + (lane & 15);
        int rb  = m16 * 16 + ((lane >> 4) << 3);
#pragma unroll
        for (int i = 0; i < 8; ++i) {
          float g, gp;
          gelu_fwd_bwd(c[i], g, gp);
          int r = rb + i;
          sHg[r * HIDN + col] = (_Float16)g;
          sHgT[col * MT + r]  = (_Float16)g;
          sGp[r * HIDN + col] = (_Float16)gp;
        }
      }
      __syncthreads();

      // Phase 3: Out = Hg @ W2 ; dOut = Out - Y (loss scale folded into update)
      for (int t = wv; t < 2 * (DD / 16); t += 8) {
        int m16 = t >> 2, ot = t & 3;
        v8f c = {};
#pragma unroll
        for (int kk = 0; kk < HIDN; kk += 32) {
          v16h a  = frag_a(sHg, HIDN, m16 * 16, kk, lane);
          v16h bm = frag_b(w2t16, HIDN, kk, ot * 16, lane);
          c = wmma_f16(a, bm, c);
        }
        int col = ot * 16 + (lane & 15);
        int rb  = m16 * 16 + ((lane >> 4) << 3);
#pragma unroll
        for (int i = 0; i < 8; ++i) {
          int r = rb + i;
          float e = c[i] - yg[r * DD + col];
          sdO[r * DD + col]  = (_Float16)e;
          sdOT[col * MT + r] = (_Float16)e;
        }
      }
      __syncthreads();

      // Phase 4a: gradW2^T = dOut^T @ Hg ; master update m2 -= GS * grad
      for (int t = wv; t < (DD / 16) * (HIDN / 16); t += 8) {
        int ot = t >> 4, et = t & 15;
        v8f c = {};
        v16h a  = frag_a(sdOT, MT, ot * 16, 0, lane);
        v16h bm = frag_b(sHgT, MT, 0, et * 16, lane);
        c = wmma_f16(a, bm, c);
        int col = et * 16 + (lane & 15);
        int rb  = ot * 16 + ((lane >> 4) << 3);
#pragma unroll
        for (int i = 0; i < 8; ++i) m2[(rb + i) * HIDN + col] -= GS * c[i];
      }

      // Phase 4b: dHpre = (dOut @ W2^T) * gelu'(Hpre), stored transposed
      for (int t = wv; t < 2 * (HIDN / 16); t += 8) {
        int m16 = t >> 4, et = t & 15;
        v8f c = {};
#pragma unroll
        for (int kk = 0; kk < DD; kk += 32) {
          v16h a  = frag_a(sdO, DD, m16 * 16, kk, lane);
          v16h bm = frag_b(w2c16, DD, kk, et * 16, lane);
          c = wmma_f16(a, bm, c);
        }
        int col = et * 16 + (lane & 15);
        int rb  = m16 * 16 + ((lane >> 4) << 3);
#pragma unroll
        for (int i = 0; i < 8; ++i) {
          int r = rb + i;
          float dh = c[i] * (float)sGp[r * HIDN + col];
          sdHT[col * MT + r] = (_Float16)dh;
        }
      }
      __syncthreads();

      // Phase 5: gradW1^T = dHpre^T @ X ; master update m1 -= GS * grad
      for (int t = wv; t < (HIDN / 16) * (DD / 16); t += 8) {
        int et = t >> 2, dt = t & 3;
        v8f c = {};
        v16h a  = frag_a(sdHT, MT, et * 16, 0, lane);
        v16h bm = frag_b(sXT, MT, 0, dt * 16, lane);
        c = wmma_f16(a, bm, c);
        int col = dt * 16 + (lane & 15);
        int rb  = et * 16 + ((lane >> 4) << 3);
#pragma unroll
        for (int i = 0; i < 8; ++i) m1[(rb + i) * DD + col] -= GS * c[i];
      }
      __threadfence_block();
      __syncthreads();
    }
  }

  // Final eval: out = gelu(q @ W1f) @ W2f for all N rows of this (b,h)
  for (int i = tid; i < HIDN * DD; i += 256) w1t16[i] = (_Float16)m1[i];
  for (int i = tid; i < DD * HIDN; i += 256) w2t16[i] = (_Float16)m2[i];
  __syncthreads();

  const float* qb_ = gq + (size_t)wg * NN * DD;
  float* ob_ = gout + (size_t)b * NN * (HH * DD) + h * DD;

#pragma unroll 1
  for (int qt = 0; qt < NN / MT; ++qt) {
    const float* xg = qb_ + qt * MT * DD;
    for (int i = tid; i < MT * DD; i += 256) {
      int m = i >> 6, d = i & (DD - 1);
      sX[m * DD + d] = (_Float16)xg[i];
    }
    if (qt + 1 < NN / MT) {
      __builtin_prefetch(xg + MT * DD + tid * 8, 0, 0);
    }
    __syncthreads();

    for (int t = wv; t < 2 * (HIDN / 16); t += 8) {
      int m16 = t >> 4, et = t & 15;
      v8f c = {};
#pragma unroll
      for (int kk = 0; kk < DD; kk += 32) {
        v16h a  = frag_a(sX, DD, m16 * 16, kk, lane);
        v16h bm = frag_b(w1t16, DD, kk, et * 16, lane);
        c = wmma_f16(a, bm, c);
      }
      int col = et * 16 + (lane & 15);
      int rb  = m16 * 16 + ((lane >> 4) << 3);
#pragma unroll
      for (int i = 0; i < 8; ++i) {
        float g, gp;
        gelu_fwd_bwd(c[i], g, gp);
        sHg[(rb + i) * HIDN + col] = (_Float16)g;
      }
    }
    __syncthreads();

    for (int t = wv; t < 2 * (DD / 16); t += 8) {
      int m16 = t >> 2, ot = t & 3;
      v8f c = {};
#pragma unroll
      for (int kk = 0; kk < HIDN; kk += 32) {
        v16h a  = frag_a(sHg, HIDN, m16 * 16, kk, lane);
        v16h bm = frag_b(w2t16, HIDN, kk, ot * 16, lane);
        c = wmma_f16(a, bm, c);
      }
      int col = ot * 16 + (lane & 15);
      int rb  = m16 * 16 + ((lane >> 4) << 3);
#pragma unroll
      for (int i = 0; i < 8; ++i) {
        int row = qt * MT + rb + i;
        ob_[(size_t)row * (HH * DD) + col] = c[i];
      }
    }
    __syncthreads();
  }
}

extern "C" void kernel_launch(void* const* d_in, const int* in_sizes, int n_in,
                              void* d_out, int out_size, void* d_ws,
                              size_t ws_size, hipStream_t stream) {
  (void)in_sizes; (void)n_in; (void)out_size; (void)ws_size;
  const float* q  = (const float*)d_in[0];
  const float* k  = (const float*)d_in[1];
  const float* v  = (const float*)d_in[2];
  const float* W1 = (const float*)d_in[3];
  const float* W2 = (const float*)d_in[4];
  float* out = (float*)d_out;
  float* ws  = (float*)d_ws;  // needs 128 * 32768 floats = 16 MB

  constexpr size_t SMEM_HALFS =
      (size_t)HIDN * DD + (size_t)DD * HIDN + (size_t)HIDN * DD +  // weights f16
      (size_t)MT * DD * 2 +                                        // X, X^T
      (size_t)MT * HIDN * 2 + (size_t)HIDN * MT +                  // Hg, Gp, Hg^T
      (size_t)MT * DD * 2 +                                        // dOut, dOut^T
      (size_t)HIDN * MT;                                           // dHpre^T
  constexpr size_t SMEM_BYTES = SMEM_HALFS * sizeof(_Float16);     // 180224 B

  hipFuncSetAttribute((const void*)ttt_mhsa_kernel,
                      hipFuncAttributeMaxDynamicSharedMemorySize,
                      (int)SMEM_BYTES);
  ttt_mhsa_kernel<<<dim3(BB * HH), dim3(256), SMEM_BYTES, stream>>>(
      q, k, v, W1, W2, out, ws);
}
